// PointerNet_54434415509912
// MI455X (gfx1250) — matrix operand order repaired and checked by reading, not compiled, not run on hardware
//
#include <hip/hip_runtime.h>

// ---------------------------------------------------------------------------
// PointerNet forward for MI455X (gfx1250).
// GEMMs: v_wmma_f32_16x16x32_bf16, one wave per 16x64 C tile (A fragment
// reused across 4 B fragments). K=256 is compile-time: the 8-chunk K loop is
// fully unrolled with an even/odd fragment double-buffer; __launch_bounds__
// lifts the VGPR cap so both fragment generations stay live (loads overlap
// WMMA instead of s_wait_loadcnt 0 before every WMMA).
// Attention / softmax / sampling: per-batch-row blocks with LDS reductions.
// ---------------------------------------------------------------------------

#define DEVINL __device__ __forceinline__

typedef __bf16 bf16_t;
typedef __attribute__((ext_vector_type(8)))  __bf16 v8bf;
typedef __attribute__((ext_vector_type(16))) __bf16 v16bf;
typedef __attribute__((ext_vector_type(8)))  float   v8f;

constexpr int kB  = 512;
constexpr int kS  = 100;
constexpr int kE  = 256;
constexpr int kH  = 256;
constexpr int kH4 = 1024;
constexpr float kCexp = 10.0f;

static_assert(kE == kH, "kernels assume E == H (256 threads cover both)");

DEVINL bf16_t f2bf(float f) {
  unsigned u = __builtin_bit_cast(unsigned, f);
  unsigned r = u + 0x7FFFu + ((u >> 16) & 1u);   // round-to-nearest-even
  unsigned short s = (unsigned short)(r >> 16);
  return __builtin_bit_cast(bf16_t, s);
}
DEVINL float sigf(float x) { return 1.0f / (1.0f + __expf(-x)); }

DEVINL unsigned hash2(unsigned a, unsigned b) {
  unsigned x = a * 0x9E3779B1u ^ (b + 0x7F4A7C15u + (a << 6) + (a >> 2));
  x ^= x >> 16; x *= 0x85EBCA6Bu;
  x ^= x >> 13; x *= 0xC2B2AE35u;
  x ^= x >> 16;
  return x;
}

// ---- WMMA fragment loaders (contiguous k per lane half, A and B alike) ---------------------------
// A (16x32 bf16): lanes 0-15 = rows M=0..15; halves of wave cover
// K=(hi?16:0)+0..15, hi = lane>=16. One 32B load per lane (matches B).
DEVINL v16bf load_a_frag(const bf16_t* __restrict__ A, int lda, int m0, int k0, int lane) {
  int m  = m0 + (lane & 15);
  int hi = (lane >> 4) & 1;
  return *reinterpret_cast<const v16bf*>(A + (size_t)m * lda + k0 + hi * 16);
}
// B (32x16 bf16): W stored row-major [N,K]; lane n = lane&15, halves of wave
// cover K=(hi?16:0)+0..15 -> one contiguous 32-byte (v16bf) load per lane.
DEVINL v16bf load_b_frag(const bf16_t* __restrict__ W, int ldw, int n0, int k0, int lane) {
  int n  = n0 + (lane & 15);
  int hi = (lane >> 4) & 1;
  return *reinterpret_cast<const v16bf*>(W + (size_t)n * ldw + k0 + hi * 16);
}
DEVINL v8f wmma_bf16(v16bf a, v16bf b, v8f c) {
  return __builtin_amdgcn_wmma_f32_16x16x32_bf16(false, a, false, b, (short)0, c, false, false);
}

// Accumulate one A-panel x W-panel pair into 4 accumulators (16x64 tile).
// KCH = number of 32-wide K chunks (compile-time -> full unroll, even/odd
// double buffer collapses to register renaming).
template <int KCH>
DEVINL void gemm_panel(const bf16_t* __restrict__ A, int lda,
                       const bf16_t* __restrict__ W, int ldw,
                       int m0, int n0, int lane, v8f acc[4]) {
  v16bf a0, a1, b0[4], b1[4];
  a0 = load_a_frag(A, lda, m0, 0, lane);
#pragma unroll
  for (int j = 0; j < 4; ++j) b0[j] = load_b_frag(W, ldw, n0 + 16 * j, 0, lane);
#pragma unroll
  for (int kc = 0; kc < KCH; ++kc) {
    const bool even = (kc & 1) == 0;
    if (kc + 2 < KCH) {
      __builtin_prefetch((const void*)(A + (size_t)(m0 + (lane & 15)) * lda + (kc + 2) * 32), 0, 3);
    }
    if (kc + 1 < KCH) {
      const int kn = (kc + 1) * 32;
      if (even) {
        a1 = load_a_frag(A, lda, m0, kn, lane);
#pragma unroll
        for (int j = 0; j < 4; ++j) b1[j] = load_b_frag(W, ldw, n0 + 16 * j, kn, lane);
      } else {
        a0 = load_a_frag(A, lda, m0, kn, lane);
#pragma unroll
        for (int j = 0; j < 4; ++j) b0[j] = load_b_frag(W, ldw, n0 + 16 * j, kn, lane);
      }
    }
    if (even) {
#pragma unroll
      for (int j = 0; j < 4; ++j) acc[j] = wmma_bf16(a0, b0[j], acc[j]);
    } else {
#pragma unroll
      for (int j = 0; j < 4; ++j) acc[j] = wmma_bf16(a1, b1[j], acc[j]);
    }
  }
}

DEVINL void store_tile(float* __restrict__ Cm, int ldc, int m0, int n0, int lane,
                       const v8f acc[4], const float* __restrict__ b1,
                       const float* __restrict__ b2) {
  int hi = (lane >> 4) & 1;
#pragma unroll
  for (int j = 0; j < 4; ++j) {
    int col  = n0 + 16 * j + (lane & 15);
    float bs = b1[col] + (b2 ? b2[col] : 0.0f);
#pragma unroll
    for (int r = 0; r < 8; ++r)
      Cm[(size_t)(m0 + r + hi * 8) * ldc + col] = acc[j][r] + bs;
  }
}

// C = A[M,K](bf16) * W[N,K]^T(bf16) + bias[N], f32 out. One wave per 16x64.
template <int K>
__global__ void __launch_bounds__(128, 1)
pn_gemm1(const bf16_t* __restrict__ A, int lda,
         const bf16_t* __restrict__ W, int ldw,
         const float* __restrict__ bias,
         float* __restrict__ Cm, int ldc, int M, int N) {
  int lane   = threadIdx.x & 31;
  int wid    = blockIdx.x * (blockDim.x >> 5) + (threadIdx.x >> 5);
  int tilesN = N >> 6;
  int tm = wid / tilesN, tn = wid % tilesN;
  if (tm >= (M >> 4)) return;               // wave-uniform: EXEC stays full
  int m0 = tm << 4, n0 = tn << 6;
  v8f acc[4] = {};
  gemm_panel<K / 32>(A, lda, W, ldw, m0, n0, lane, acc);
  store_tile(Cm, ldc, m0, n0, lane, acc, bias, nullptr);
}

// C = A1*W1^T + A2*W2^T + b1 + b2  (LSTM gate pre-activations, N = 4H)
template <int K1, int K2>
__global__ void __launch_bounds__(128, 1)
pn_gemm2(const bf16_t* __restrict__ A1, int lda1,
         const bf16_t* __restrict__ W1, int ldw1,
         const bf16_t* __restrict__ A2, int lda2,
         const bf16_t* __restrict__ W2, int ldw2,
         const float* __restrict__ b1, const float* __restrict__ b2,
         float* __restrict__ Cm, int ldc, int M, int N) {
  int lane   = threadIdx.x & 31;
  int wid    = blockIdx.x * (blockDim.x >> 5) + (threadIdx.x >> 5);
  int tilesN = N >> 6;
  int tm = wid / tilesN, tn = wid % tilesN;
  if (tm >= (M >> 4)) return;
  int m0 = tm << 4, n0 = tn << 6;
  v8f acc[4] = {};
  gemm_panel<K1 / 32>(A1, lda1, W1, ldw1, m0, n0, lane, acc);
  gemm_panel<K2 / 32>(A2, lda2, W2, ldw2, m0, n0, lane, acc);
  store_tile(Cm, ldc, m0, n0, lane, acc, b1, b2);
}

// ---- elementwise / setup kernels ------------------------------------------
__global__ void pn_zero_bytes(unsigned char* p, int n) {
  int i = blockIdx.x * blockDim.x + threadIdx.x;
  if (i < n) p[i] = 0;
}
__global__ void pn_cast_bf16(const float* __restrict__ x, bf16_t* __restrict__ y, int n) {
  int i = blockIdx.x * blockDim.x + threadIdx.x;
  if (i < n) y[i] = f2bf(x[i]);
}
// embedded[s][b][e] = inputs[b,0,s]*emb[0,e] + inputs[b,1,s]*emb[1,e]
__global__ void pn_embed(const float* __restrict__ inputs, const float* __restrict__ emb,
                         bf16_t* __restrict__ out) {
  int i = blockIdx.x * blockDim.x + threadIdx.x;
  if (i >= kS * kB * kE) return;
  int e = i % kE, b = (i / kE) % kB, s = i / (kE * kB);
  float v = inputs[(size_t)b * 2 * kS + s] * emb[e]
          + inputs[(size_t)b * 2 * kS + kS + s] * emb[kE + e];
  out[i] = f2bf(v);
}
__global__ void pn_bcast_start(const float* __restrict__ start, bf16_t* __restrict__ dec_in) {
  int i = blockIdx.x * blockDim.x + threadIdx.x;
  if (i >= kB * kE) return;
  dec_in[i] = f2bf(start[i % kE]);
}
// LSTM pointwise: torch gate order i,f,g,o
__global__ void pn_lstm_update(const float* __restrict__ G, float* __restrict__ c,
                               bf16_t* __restrict__ h_bf,
                               float* __restrict__ enc_f, bf16_t* __restrict__ enc_b,
                               int store) {
  int i = blockIdx.x * blockDim.x + threadIdx.x;
  if (i >= kB * kH) return;
  int b = i / kH, j = i % kH;
  const float* g = G + (size_t)b * kH4;
  float ig = sigf(g[j]);
  float fg = sigf(g[kH + j]);
  float gg = tanhf(g[2 * kH + j]);
  float og = sigf(g[3 * kH + j]);
  float cn = fg * c[i] + ig * gg;
  float hn = og * tanhf(cn);
  c[i] = cn;
  h_bf[i] = f2bf(hn);
  if (store) { enc_f[i] = hn; enc_b[i] = f2bf(hn); }
}

DEVINL float block_reduce_sum(float v, float* red) {
  int t = threadIdx.x;
  red[t] = v; __syncthreads();
  for (int off = 128; off > 0; off >>= 1) {
    if (t < off) red[t] += red[t + off];
    __syncthreads();
  }
  float r = red[0]; __syncthreads();
  return r;
}
DEVINL float block_reduce_max(float v, float* red) {
  int t = threadIdx.x;
  red[t] = v; __syncthreads();
  for (int off = 128; off > 0; off >>= 1) {
    if (t < off) red[t] = fmaxf(red[t], red[t + off]);
    __syncthreads();
  }
  float r = red[0]; __syncthreads();
  return r;
}
// Parallel in-place softmax over arr[0..kS-1] (blockDim.x == 256 threads).
DEVINL void block_softmax(float* arr, float* red) {
  int t = threadIdx.x;
  float v  = (t < kS) ? arr[t] : -INFINITY;
  float mx = block_reduce_max(v, red);
  float e  = (t < kS) ? __expf(arr[t] - mx) : 0.0f;
  float se = block_reduce_sum(e, red);
  if (t < kS) arr[t] = e / se;
  __syncthreads();
}

// glimpse: mask update, lg = V.tanh(qq+ref_g) (masked), softmax, q = sum p*enc
__global__ void pn_glimpse(int step, const float* __restrict__ qq,
                           const float* __restrict__ ref_g, const float* __restrict__ gV,
                           const float* __restrict__ enc, unsigned char* __restrict__ mask,
                           const int* __restrict__ idx_prev, bf16_t* __restrict__ q_bf) {
  int b = blockIdx.x, t = threadIdx.x;
  __shared__ float red[256];
  __shared__ float lg[128];
  if (step > 0 && t == 0) mask[b * kS + idx_prev[b]] = 1;
  __syncthreads();
  float qv = qq[(size_t)b * kH + t];
  float gv = gV[t];
  for (int s = 0; s < kS; ++s) {
    float v   = gv * tanhf(qv + ref_g[((size_t)s * kB + b) * kH + t]);
    float sum = block_reduce_sum(v, red);
    if (t == 0) lg[s] = mask[b * kS + s] ? -INFINITY : sum;
  }
  __syncthreads();
  block_softmax(lg, red);
  float acc = 0.f;
  for (int s = 0; s < kS; ++s) acc += lg[s] * enc[((size_t)s * kB + b) * kH + t];
  q_bf[(size_t)b * kH + t] = f2bf(acc);
}

// pointer: lp = C*tanh(V.tanh(qq+ref_p)) (masked), softmax -> probs, sample,
// gather next decoder input from embedded.
__global__ void pn_pointer(int step, const float* __restrict__ qq,
                           const float* __restrict__ ref_p, const float* __restrict__ pV,
                           const unsigned char* __restrict__ mask, int* __restrict__ idx,
                           const bf16_t* __restrict__ embedded, bf16_t* __restrict__ dec_in,
                           float* __restrict__ out_probs, float* __restrict__ out_idx) {
  int b = blockIdx.x, t = threadIdx.x;
  __shared__ float red[256];
  __shared__ float lp[128];
  __shared__ int   sidx;
  float qv = qq[(size_t)b * kH + t];
  float pv = pV[t];
  for (int s = 0; s < kS; ++s) {
    float v   = pv * tanhf(qv + ref_p[((size_t)s * kB + b) * kH + t]);
    float sum = block_reduce_sum(v, red);
    if (t == 0) lp[s] = mask[b * kS + s] ? -INFINITY : kCexp * tanhf(sum);
  }
  __syncthreads();
  block_softmax(lp, red);
  if (t == 0) {
    float u = (float)(hash2((unsigned)step, (unsigned)b) >> 8) * (1.0f / 16777216.0f);
    float cum = 0.f; int pick = kS - 1;
    for (int s = 0; s < kS; ++s) { cum += lp[s]; if (u < cum) { pick = s; break; } }
    sidx = pick;
    idx[b] = pick;
    out_idx[(size_t)step * kB + b] = (float)pick;
  }
  __syncthreads();
  for (int s = t; s < kS; s += 256)
    out_probs[((size_t)step * kB + b) * kS + s] = lp[s];
  int pick = sidx;
  dec_in[(size_t)b * kE + t] = embedded[((size_t)pick * kB + b) * kE + t];
}

// ---------------------------------------------------------------------------
extern "C" void kernel_launch(void* const* d_in, const int* in_sizes, int n_in,
                              void* d_out, int out_size, void* d_ws, size_t ws_size,
                              hipStream_t stream) {
  const float* inputs  = (const float*)d_in[0];
  const float* emb     = (const float*)d_in[1];
  const float* encWih  = (const float*)d_in[2];
  const float* encWhh  = (const float*)d_in[3];
  const float* enc_bih = (const float*)d_in[4];
  const float* enc_bhh = (const float*)d_in[5];
  const float* decWih  = (const float*)d_in[6];
  const float* decWhh  = (const float*)d_in[7];
  const float* dec_bih = (const float*)d_in[8];
  const float* dec_bhh = (const float*)d_in[9];
  const float* gWq     = (const float*)d_in[10];
  const float* g_bq    = (const float*)d_in[11];
  const float* gWr     = (const float*)d_in[12];
  const float* g_br    = (const float*)d_in[13];
  const float* g_V     = (const float*)d_in[14];
  const float* pWq     = (const float*)d_in[15];
  const float* p_bq    = (const float*)d_in[16];
  const float* pWr     = (const float*)d_in[17];
  const float* p_br    = (const float*)d_in[18];
  const float* p_V     = (const float*)d_in[19];
  const float* start   = (const float*)d_in[20];

  float* out_probs = (float*)d_out;                              // [S,B,S]
  float* out_idx   = (float*)d_out + (size_t)kS * kB * kS;       // [S,B]

  // ---- workspace carve-out ----
  char* cur = (char*)d_ws;
  auto take = [&](size_t bytes) -> void* {
    void* p = (void*)cur;
    cur += (bytes + 255) & ~(size_t)255;
    return p;
  };
  bf16_t* embedded_bf = (bf16_t*)take((size_t)kS * kB * kE * 2);
  float*  enc_f       = (float*) take((size_t)kS * kB * kH * 4);
  bf16_t* enc_b       = (bf16_t*)take((size_t)kS * kB * kH * 2);
  float*  ref_g       = (float*) take((size_t)kS * kB * kH * 4);
  float*  ref_p       = (float*) take((size_t)kS * kB * kH * 4);
  float*  G           = (float*) take((size_t)kB * kH4 * 4);
  float*  c           = (float*) take((size_t)kB * kH * 4);
  bf16_t* h_bf        = (bf16_t*)take((size_t)kB * kH * 2);
  float*  qq          = (float*) take((size_t)kB * kH * 4);
  bf16_t* q_bf        = (bf16_t*)take((size_t)kB * kH * 2);
  bf16_t* dec_in      = (bf16_t*)take((size_t)kB * kE * 2);
  unsigned char* mask = (unsigned char*)take((size_t)kB * kS);
  int*    idx         = (int*)   take((size_t)kB * 4);
  bf16_t* encWih_bf   = (bf16_t*)take((size_t)kH4 * kE * 2);
  bf16_t* encWhh_bf   = (bf16_t*)take((size_t)kH4 * kH * 2);
  bf16_t* decWih_bf   = (bf16_t*)take((size_t)kH4 * kE * 2);
  bf16_t* decWhh_bf   = (bf16_t*)take((size_t)kH4 * kH * 2);
  bf16_t* gWq_bf      = (bf16_t*)take((size_t)kH * kH * 2);
  bf16_t* gWr_bf      = (bf16_t*)take((size_t)kH * kH * 2);
  bf16_t* pWq_bf      = (bf16_t*)take((size_t)kH * kH * 2);
  bf16_t* pWr_bf      = (bf16_t*)take((size_t)kH * kH * 2);
  if ((size_t)(cur - (char*)d_ws) > ws_size) return;  // not enough scratch

  const int T256 = 256;
  auto gridFor = [](int n) { return (n + 255) / 256; };

  // ---- weight casts (L2-resident bf16 copies) ----
  pn_cast_bf16<<<gridFor(kH4 * kE), T256, 0, stream>>>(encWih, encWih_bf, kH4 * kE);
  pn_cast_bf16<<<gridFor(kH4 * kH), T256, 0, stream>>>(encWhh, encWhh_bf, kH4 * kH);
  pn_cast_bf16<<<gridFor(kH4 * kE), T256, 0, stream>>>(decWih, decWih_bf, kH4 * kE);
  pn_cast_bf16<<<gridFor(kH4 * kH), T256, 0, stream>>>(decWhh, decWhh_bf, kH4 * kH);
  pn_cast_bf16<<<gridFor(kH * kH),  T256, 0, stream>>>(gWq, gWq_bf, kH * kH);
  pn_cast_bf16<<<gridFor(kH * kH),  T256, 0, stream>>>(gWr, gWr_bf, kH * kH);
  pn_cast_bf16<<<gridFor(kH * kH),  T256, 0, stream>>>(pWq, pWq_bf, kH * kH);
  pn_cast_bf16<<<gridFor(kH * kH),  T256, 0, stream>>>(pWr, pWr_bf, kH * kH);

  // ---- state init ----
  pn_zero_bytes<<<gridFor(kB * kH * 4), T256, 0, stream>>>((unsigned char*)c, kB * kH * 4);
  pn_zero_bytes<<<gridFor(kB * kH * 2), T256, 0, stream>>>((unsigned char*)h_bf, kB * kH * 2);
  pn_zero_bytes<<<gridFor(kB * kS),     T256, 0, stream>>>(mask, kB * kS);

  // ---- embedding + decoder start input ----
  pn_embed<<<gridFor(kS * kB * kE), T256, 0, stream>>>(inputs, emb, embedded_bf);
  pn_bcast_start<<<gridFor(kB * kE), T256, 0, stream>>>(start, dec_in);

  // ---- encoder: 100 sequential LSTM steps (WMMA GEMM + pointwise) ----
  const int gateBlocks = (kB / 16) * (kH4 / 64) / 4;   // 128 blocks x 4 waves
  for (int s = 0; s < kS; ++s) {
    pn_gemm2<kE, kH><<<gateBlocks, 128, 0, stream>>>(
        embedded_bf + (size_t)s * kB * kE, kE, encWih_bf, kE,
        h_bf, kH, encWhh_bf, kH,
        enc_bih, enc_bhh, G, kH4, kB, kH4);
    pn_lstm_update<<<gridFor(kB * kH), T256, 0, stream>>>(
        G, c, h_bf, enc_f + (size_t)s * kB * kH, enc_b + (size_t)s * kB * kH, 1);
  }

  // ---- step-invariant reference projections (big parallel WMMA GEMMs) ----
  const int refBlocks = ((kS * kB) / 16) * (kH / 64) / 4;  // 3200
  pn_gemm1<kH><<<refBlocks, 128, 0, stream>>>(enc_b, kH, gWr_bf, kH, g_br, ref_g, kH, kS * kB, kH);
  pn_gemm1<kH><<<refBlocks, 128, 0, stream>>>(enc_b, kH, pWr_bf, kH, p_br, ref_p, kH, kS * kB, kH);

  // ---- decoder: 100 sequential steps ----
  const int qBlocks = (kB / 16) * (kH / 64) / 4;           // 32
  for (int step = 0; step < kS; ++step) {
    pn_gemm2<kE, kH><<<gateBlocks, 128, 0, stream>>>(
        dec_in, kE, decWih_bf, kE,
        h_bf, kH, decWhh_bf, kH,
        dec_bih, dec_bhh, G, kH4, kB, kH4);
    pn_lstm_update<<<gridFor(kB * kH), T256, 0, stream>>>(G, c, h_bf, nullptr, nullptr, 0);

    pn_gemm1<kH><<<qBlocks, 128, 0, stream>>>(h_bf, kH, gWq_bf, kH, g_bq, qq, kH, kB, kH);
    pn_glimpse<<<kB, T256, 0, stream>>>(step, qq, ref_g, g_V, enc_f, mask, idx, q_bf);

    pn_gemm1<kH><<<qBlocks, 128, 0, stream>>>(q_bf, kH, pWq_bf, kH, p_bq, qq, kH, kB, kH);
    pn_pointer<<<kB, T256, 0, stream>>>(step, qq, ref_p, p_V, mask, idx,
                                        embedded_bf, dec_in, out_probs, out_idx);
  }
}
